// RobustFill_62397284876931
// MI455X (gfx1250) — matrix operand order repaired and checked by reading, not compiled, not run on hardware
//
#include <hip/hip_runtime.h>
#include <stdint.h>
#include <math.h>

// ---------------------------------------------------------------------------
// RobustFill LSTM stack on MI455X (gfx1250).
//
// Strategy (see analysis): 524 strictly-sequential matvec steps, 1 MAC/elem
// => purely bandwidth bound. BF16 weights (one-time convert) shrink the whole
// model to 113 MB < 192 MB L2, so every step after the first runs from L2.
// Matvec is mapped onto v_wmma_f32_16x16x32_bf16 with the vector chunk
// broadcast into all 16 B-columns (layout-robust; fp32 accumulate).
// Fallback path (small workspace): v_wmma_f32_16x16x4_f32 on fp32 weights.
// ---------------------------------------------------------------------------

#define HID   2048
#define STR   512
#define GSZ   (4 * HID)      // 8192 gate rows
#define LSEQ  256
#define PSTEP 10
#define NTOK  128

typedef __attribute__((ext_vector_type(16))) __bf16 bf16x16;
typedef __attribute__((ext_vector_type(8)))  float  f32x8;
typedef __attribute__((ext_vector_type(2)))  float  f32x2;

static __device__ __forceinline__ uint16_t f32_to_bf16(float f) {
  uint32_t u = __float_as_uint(f);
  u += 0x7FFFu + ((u >> 16) & 1u);      // round-to-nearest-even
  return (uint16_t)(u >> 16);
}

union BfVec { bf16x16 v; uint4 q[2]; };

// ---------------------------------------------------------------------------
// BF16 matvec accumulate: acc (16x16 f32 C/D tile, all columns equal) +=
// W[row0:row0+16, :K] * x[:K], using 16x16x32 bf16 WMMA.
// Per 16-bit A-matrix layout: lanes 0-15 hold K 0-7 & 16-23, lanes 16-31 hold
// K 8-15 & 24-31 of their row.  B: vector chunk replicated into all columns
// (same K order as A: half-wave 0 holds K 0-7 & 16-23, half-wave 1 K 8-15 & 24-31).
// ---------------------------------------------------------------------------
static __device__ __forceinline__ void mv_acc_bf16(
    f32x8& acc, const uint16_t* __restrict__ W, int K,
    const uint16_t* __restrict__ x, int row, int half) {
  const size_t rowoff = (size_t)row * (size_t)K;
  for (int k0 = 0; k0 < K; k0 += 32) {
    BfVec a, b;
    a.q[0] = *(const uint4*)(W + rowoff + (size_t)(k0 + half * 8));
    a.q[1] = *(const uint4*)(W + rowoff + (size_t)(k0 + 16 + half * 8));
    b.q[0] = *(const uint4*)(x + (k0 + half * 8));
    b.q[1] = *(const uint4*)(x + (k0 + 16 + half * 8));
    __builtin_prefetch(W + rowoff + (size_t)(k0 + 128), 0, 1);
    acc = __builtin_amdgcn_wmma_f32_16x16x32_bf16(
        false, a.v, false, b.v, (short)0, acc, false, false);
  }
}

// FP32 fallback: 16x16x4 f32 WMMA.  A layout: lanes 0-15 hold K0,K1; lanes
// 16-31 hold K2,K3.  B replicated into all columns the same way.
static __device__ __forceinline__ void mv_acc_f32(
    f32x8& acc, const float* __restrict__ W, int K,
    const float* __restrict__ x, int row, int half) {
  const size_t rowoff = (size_t)row * (size_t)K;
  for (int k0 = 0; k0 < K; k0 += 4) {
    f32x2 a = *(const f32x2*)(W + rowoff + (size_t)(k0 + half * 2));
    f32x2 b = *(const f32x2*)(x + (k0 + half * 2));
    __builtin_prefetch(W + rowoff + (size_t)(k0 + 64), 0, 1);
    acc = __builtin_amdgcn_wmma_f32_16x16x4_f32(
        false, a, false, b, (short)0, acc, false, false);
  }
}

// ---------------------------------------------------------------------------
// One LSTM-step matvec: g = W1*x1 (+ W2*x2) + bias.  512 waves, 16 rows each.
// x1 is either selected from an embedding table via tokens[t] / eos, or used
// directly (tokens == nullptr).
// ---------------------------------------------------------------------------
__global__ void __launch_bounds__(256)
mv_step_bf16(const uint16_t* __restrict__ W1, int K1,
             const uint16_t* __restrict__ xtab,
             const int* __restrict__ tokens, int t, int tlen,
             const uint16_t* __restrict__ eos,
             const uint16_t* __restrict__ W2, int K2,
             const uint16_t* __restrict__ x2,
             const float* __restrict__ bias,
             float* __restrict__ g) {
  const int lane = threadIdx.x & 31;
  const int half = lane >> 4;
  const int m    = lane & 15;
  const int wave = (blockIdx.x * blockDim.x + threadIdx.x) >> 5;
  const int row0 = wave * 16;
  const int row  = row0 + m;

  const uint16_t* x1 = xtab;
  if (tokens)
    x1 = (t < tlen) ? (xtab + (size_t)tokens[t] * (size_t)K1) : eos;

  f32x8 acc = {};
  mv_acc_bf16(acc, W1, K1, x1, row, half);
  if (W2) mv_acc_bf16(acc, W2, K2, x2, row, half);

  if (m == 0) {  // lanes 0 and 16 hold rows 0-7 / 8-15 of the tile
    const int rb = row0 + half * 8;
#pragma unroll
    for (int j = 0; j < 8; ++j) g[rb + j] = acc[j] + bias[rb + j];
  }
}

__global__ void __launch_bounds__(256)
mv_step_f32(const float* __restrict__ W1, int K1,
            const float* __restrict__ xtab,
            const int* __restrict__ tokens, int t, int tlen,
            const float* __restrict__ eos,
            const float* __restrict__ W2, int K2,
            const float* __restrict__ x2,
            const float* __restrict__ bias,
            float* __restrict__ g) {
  const int lane = threadIdx.x & 31;
  const int half = lane >> 4;
  const int m    = lane & 15;
  const int wave = (blockIdx.x * blockDim.x + threadIdx.x) >> 5;
  const int row0 = wave * 16;
  const int row  = row0 + m;

  const float* x1 = xtab;
  if (tokens)
    x1 = (t < tlen) ? (xtab + (size_t)tokens[t] * (size_t)K1) : eos;

  f32x8 acc = {};
  mv_acc_f32(acc, W1, K1, x1, row, half);
  if (W2) mv_acc_f32(acc, W2, K2, x2, row, half);

  if (m == 0) {
    const int rb = row0 + half * 8;
#pragma unroll
    for (int j = 0; j < 8; ++j) g[rb + j] = acc[j] + bias[rb + j];
  }
}

// ---------------------------------------------------------------------------
// Elementwise LSTM cell: gate order i, f, g, o (PyTorch convention).
// ---------------------------------------------------------------------------
__global__ void lstm_gates(const float* __restrict__ g,
                           float* __restrict__ c,
                           float* __restrict__ h,
                           uint16_t* __restrict__ hbf,
                           float* __restrict__ out) {
  const int j = blockIdx.x * blockDim.x + threadIdx.x;
  if (j >= HID) return;
  const float gi = g[j];
  const float gf = g[HID + j];
  const float gg = g[2 * HID + j];
  const float go = g[3 * HID + j];
  const float si = 1.0f / (1.0f + expf(-gi));
  const float sf = 1.0f / (1.0f + expf(-gf));
  const float so = 1.0f / (1.0f + expf(-go));
  const float c2 = sf * c[j] + si * tanhf(gg);
  const float h2 = so * tanhf(c2);
  c[j] = c2;
  h[j] = h2;
  if (hbf) hbf[j] = f32_to_bf16(h2);
  if (out) out[j] = h2;
}

__global__ void init_state(float* c, float* h, uint16_t* hbf) {
  const int j = blockIdx.x * blockDim.x + threadIdx.x;
  if (j < HID) { c[j] = 0.0f; h[j] = 0.0f; hbf[j] = 0; }
}

__global__ void cvt_bf16(const float* __restrict__ src,
                         uint16_t* __restrict__ dst, int n) {
  for (int i = blockIdx.x * blockDim.x + threadIdx.x; i < n;
       i += gridDim.x * blockDim.x)
    dst[i] = f32_to_bf16(src[i]);
}

// progW = bf16(prog_Wih + prog_Whh): both multiply the same h, so fuse once.
__global__ void add_cvt_bf16(const float* __restrict__ a,
                             const float* __restrict__ b,
                             uint16_t* __restrict__ dst, int n) {
  for (int i = blockIdx.x * blockDim.x + threadIdx.x; i < n;
       i += gridDim.x * blockDim.x)
    dst[i] = f32_to_bf16(a[i] + b[i]);
}

// ---------------------------------------------------------------------------
extern "C" void kernel_launch(void* const* d_in, const int* in_sizes, int n_in,
                              void* d_out, int out_size, void* d_ws,
                              size_t ws_size, hipStream_t stream) {
  const int*   in_seq  = (const int*)d_in[0];
  const int*   out_seq = (const int*)d_in[1];
  const float* embed   = (const float*)d_in[2];
  const float* eos     = (const float*)d_in[3];
  const float* in_Wih  = (const float*)d_in[4];
  const float* in_Whh  = (const float*)d_in[5];
  const float* in_b    = (const float*)d_in[6];
  const float* out_Wih = (const float*)d_in[7];
  const float* out_Whh = (const float*)d_in[8];
  const float* out_b   = (const float*)d_in[9];
  const float* pr_Wih  = (const float*)d_in[10];
  const float* pr_Whh  = (const float*)d_in[11];
  const float* pr_b    = (const float*)d_in[12];
  float* out = (float*)d_out;

  uint8_t* ws = (uint8_t*)d_ws;
  size_t off = 0;
  auto alloc = [&](size_t bytes) -> void* {
    void* p = ws + off;
    off += (bytes + 255) & ~(size_t)255;
    return p;
  };

  float*    g   = (float*)alloc(GSZ * sizeof(float));
  float*    c   = (float*)alloc(HID * sizeof(float));
  float*    h   = (float*)alloc(HID * sizeof(float));
  uint16_t* hbf = (uint16_t*)alloc(HID * sizeof(uint16_t));

  const size_t bf_bytes =
      ((size_t)NTOK * STR + STR +                      // embed + eos
       (size_t)GSZ * STR + (size_t)GSZ * HID +         // in_Wih + in_Whh
       (size_t)GSZ * STR + (size_t)GSZ * HID +         // out_Wih + out_Whh
       (size_t)GSZ * HID) * 2 +                        // progW (Wih+Whh fused)
      8 * 256;                                         // alignment slack
  const bool use_bf16 = (ws_size >= off + bf_bytes);   // launch-time constant

  uint16_t *embed_bf = nullptr, *eos_bf = nullptr;
  uint16_t *inWih_bf = nullptr, *inWhh_bf = nullptr;
  uint16_t *outWih_bf = nullptr, *outWhh_bf = nullptr, *progW_bf = nullptr;

  init_state<<<(HID + 255) / 256, 256, 0, stream>>>(c, h, hbf);

  if (use_bf16) {
    embed_bf  = (uint16_t*)alloc((size_t)NTOK * STR * 2);
    eos_bf    = (uint16_t*)alloc((size_t)STR * 2);
    inWih_bf  = (uint16_t*)alloc((size_t)GSZ * STR * 2);
    inWhh_bf  = (uint16_t*)alloc((size_t)GSZ * HID * 2);
    outWih_bf = (uint16_t*)alloc((size_t)GSZ * STR * 2);
    outWhh_bf = (uint16_t*)alloc((size_t)GSZ * HID * 2);
    progW_bf  = (uint16_t*)alloc((size_t)GSZ * HID * 2);

    auto cvt = [&](const float* s, uint16_t* d, int n) {
      int blocks = (n + 255) / 256;
      if (blocks > 4096) blocks = 4096;
      cvt_bf16<<<blocks, 256, 0, stream>>>(s, d, n);
    };
    cvt(embed, embed_bf, NTOK * STR);
    cvt(eos, eos_bf, STR);
    cvt(in_Wih, inWih_bf, GSZ * STR);
    cvt(in_Whh, inWhh_bf, GSZ * HID);
    cvt(out_Wih, outWih_bf, GSZ * STR);
    cvt(out_Whh, outWhh_bf, GSZ * HID);
    add_cvt_bf16<<<4096, 256, 0, stream>>>(pr_Wih, pr_Whh, progW_bf, GSZ * HID);
  }

  const dim3 mvG(64), mvB(256);            // 512 waves -> 8192 rows
  const dim3 gtG((HID + 255) / 256), gtB(256);

  // ---- input encoder: 256 tokens + eos ----
  for (int t = 0; t <= LSEQ; ++t) {
    if (use_bf16)
      mv_step_bf16<<<mvG, mvB, 0, stream>>>(inWih_bf, STR, embed_bf, in_seq, t,
                                            LSEQ, eos_bf, inWhh_bf, HID, hbf,
                                            in_b, g);
    else
      mv_step_f32<<<mvG, mvB, 0, stream>>>(in_Wih, STR, embed, in_seq, t, LSEQ,
                                           eos, in_Whh, HID, h, in_b, g);
    lstm_gates<<<gtG, gtB, 0, stream>>>(g, c, h, use_bf16 ? hbf : nullptr,
                                        nullptr);
  }

  // ---- output encoder: 256 tokens + eos (state carried over) ----
  for (int t = 0; t <= LSEQ; ++t) {
    if (use_bf16)
      mv_step_bf16<<<mvG, mvB, 0, stream>>>(outWih_bf, STR, embed_bf, out_seq,
                                            t, LSEQ, eos_bf, outWhh_bf, HID,
                                            hbf, out_b, g);
    else
      mv_step_f32<<<mvG, mvB, 0, stream>>>(out_Wih, STR, embed, out_seq, t,
                                           LSEQ, eos, out_Whh, HID, h, out_b,
                                           g);
    lstm_gates<<<gtG, gtB, 0, stream>>>(g, c, h, use_bf16 ? hbf : nullptr,
                                        nullptr);
  }

  // ---- program decoder: 10 autoregressive steps, g = (Wih+Whh)*h + b ----
  for (int s = 0; s < PSTEP; ++s) {
    if (use_bf16)
      mv_step_bf16<<<mvG, mvB, 0, stream>>>(progW_bf, HID, hbf, nullptr, 0, 0,
                                            nullptr, nullptr, 0, nullptr, pr_b,
                                            g);
    else
      mv_step_f32<<<mvG, mvB, 0, stream>>>(pr_Wih, HID, h, nullptr, 0, 0,
                                           nullptr, pr_Whh, HID, h, pr_b, g);
    lstm_gates<<<gtG, gtB, 0, stream>>>(g, c, h, use_bf16 ? hbf : nullptr,
                                        out + (size_t)s * HID);
  }
}
